// HybridChaosAwareNetwork_77446850282070
// MI455X (gfx1250) — matrix operand order repaired and checked by reading, not compiled, not run on hardware
//
#include <hip/hip_runtime.h>
#include <math.h>

// ---------------------------------------------------------------------------
// Problem constants (from the reference)
// ---------------------------------------------------------------------------
#define B_   8
#define T_   1024
#define DIN_ 64
#define R_   1024
#define E_   512
#define H_   8
#define L_   4
#define P_   3
#define DH_  64
#define LEAK_ 0.3f

typedef __bf16 bf16;
typedef bf16 v16bf __attribute__((ext_vector_type(16)));
typedef bf16 v8bf  __attribute__((ext_vector_type(8)));
typedef float v8f  __attribute__((ext_vector_type(8)));
typedef unsigned int v4u __attribute__((ext_vector_type(4)));
typedef int v8i __attribute__((ext_vector_type(8)));
typedef int v4i __attribute__((ext_vector_type(4)));

__device__ __forceinline__ v8f wmma_bf16x32(v16bf a, v16bf b, v8f c) {
  // D = A(16x32 bf16) * B(32x16 bf16) + C(16x16 f32), wave32
  return __builtin_amdgcn_wmma_f32_16x16x32_bf16(
      /*neg_a=*/false, a, /*neg_b=*/false, b,
      /*c_mod=*/(short)0, c, /*reuse_a=*/false, /*reuse_b=*/false);
}

__device__ __forceinline__ float apply_act(float v, int act) {
  if (act == 1) return tanhf(v);
  if (act == 2) return 0.5f * v * (1.0f + erff(v * 0.70710678118654752f)); // exact gelu
  if (act == 3) return fmaxf(v, 0.0f);
  return v;
}

// ---------------------------------------------------------------------------
// Elementwise helpers
// ---------------------------------------------------------------------------
__global__ void k_f32_to_bf16(const float* __restrict__ x, bf16* __restrict__ y, int n) {
  int i = blockIdx.x * blockDim.x + threadIdx.x;
  if (i < n) y[i] = (bf16)x[i];
}

__global__ void k_fill(float* __restrict__ p, int n, float v) {
  int i = blockIdx.x * blockDim.x + threadIdx.x;
  if (i < n) p[i] = v;
}

__global__ void k_fill_bf16(bf16* __restrict__ p, int n, float v) {
  int i = blockIdx.x * blockDim.x + threadIdx.x;
  if (i < n) p[i] = (bf16)v;
}

__global__ void k_add(float* __restrict__ a, const float* __restrict__ x, int n) {
  int i = blockIdx.x * blockDim.x + threadIdx.x;
  if (i < n) a[i] += x[i];
}

__global__ void k_copy(const float* __restrict__ s, float* __restrict__ d, int n) {
  int i = blockIdx.x * blockDim.x + threadIdx.x;
  if (i < n) d[i] = s[i];
}

// ---------------------------------------------------------------------------
// Generic batched WMMA GEMM:  C[z] = act( A[z] @ Bt[z]^T + bias )
//   A  : [M,K] bf16 row-major        (M must be a multiple of 32)
//   Bt : [N,K] bf16 row-major        (torch-style weight [out,in]; any N)
//   C  : [M,N] f32 or bf16
// Block = 256 threads = 8 waves arranged 4(M) x 2(N).
// Each wave computes a 32x64 sub-tile = 2x4 WMMA accumulators -> block tile
// 128x128.  A fragment reused by 4 WMMAs, B fragment by 2.  Inner loop is
// branch-free: OOB B rows are clamped to row N-1 and masked at the store.
// ---------------------------------------------------------------------------
__global__ void __launch_bounds__(256) k_gemm(
    const bf16* __restrict__ A, const bf16* __restrict__ Bt,
    const float* __restrict__ bias, void* __restrict__ Cv,
    int M, int N, int K,
    long long sA, long long sB, long long sC,
    int act, int out_bf16)
{
  const int lane = threadIdx.x & 31;
  const int wave = threadIdx.x >> 5;
  const int half = lane >> 4;     // 0: lanes 0-15, 1: lanes 16-31
  const int lr   = lane & 15;
  const int m0 = blockIdx.y * 128 + (wave >> 1) * 32;
  const int n0 = blockIdx.x * 128 + (wave & 1) * 64;
  if (m0 >= M) return;                          // wave-uniform
  const long long zb = blockIdx.z;
  A  += zb * sA;
  Bt += zb * sB;

  const bf16* arow0 = A + (long long)(m0 + lr) * K;
  const bf16* arow1 = A + (long long)(m0 + 16 + lr) * K;
  const bf16* brow[4];
#pragma unroll
  for (int j = 0; j < 4; j++) {
    int bn = n0 + j * 16 + lr;
    if (bn > N - 1) bn = N - 1;                 // clamp, store is masked
    brow[j] = Bt + (long long)bn * K;
  }

  v8f acc[2][4];
#pragma unroll
  for (int i = 0; i < 2; i++)
#pragma unroll
    for (int j = 0; j < 4; j++) { v8f z = {}; acc[i][j] = z; }

  for (int k = 0; k < K; k += 32) {
    // A 16x32 bf16 layout: lanes0-15 K in {0..15}; lanes16-31 {16..31}
    v8bf lo0 = *(const v8bf*)(arow0 + k + half * 16);
    v8bf hi0 = *(const v8bf*)(arow0 + k + half * 16 + 8);
    v8bf lo1 = *(const v8bf*)(arow1 + k + half * 16);
    v8bf hi1 = *(const v8bf*)(arow1 + k + half * 16 + 8);
    v16bf a0, a1;
#pragma unroll
    for (int i = 0; i < 8; i++) {
      a0[i] = lo0[i]; a0[i + 8] = hi0[i];
      a1[i] = lo1[i]; a1[i + 8] = hi1[i];
    }
    // B 32x16 bf16: lane's column, contiguous K run (half*16 .. half*16+15)
    v16bf bv[4];
#pragma unroll
    for (int j = 0; j < 4; j++) bv[j] = *(const v16bf*)(brow[j] + k + half * 16);
    if (k + 32 < K) __builtin_prefetch(arow0 + k + 32, 0, 1);
#pragma unroll
    for (int j = 0; j < 4; j++) {
      acc[0][j] = wmma_bf16x32(a0, bv[j], acc[0][j]);
      acc[1][j] = wmma_bf16x32(a1, bv[j], acc[1][j]);
    }
  }

  const long long cb = zb * sC;
#pragma unroll
  for (int i = 0; i < 2; i++) {
#pragma unroll
    for (int j = 0; j < 4; j++) {
      const int n = n0 + j * 16 + lr;
      if (n < N) {
        const float bsum = bias ? bias[n] : 0.0f;
#pragma unroll
        for (int r = 0; r < 8; r++) {
          const int m = m0 + i * 16 + half * 8 + r;   // C layout: VGPR r -> M
          const float v = apply_act(acc[i][j][r] + bsum, act);
          const long long idx = cb + (long long)m * N + n;
          if (out_bf16) ((bf16*)Cv)[idx] = (bf16)v;
          else          ((float*)Cv)[idx] = v;
        }
      }
    }
  }
}

// ---------------------------------------------------------------------------
// Reservoir step:  new = tanh(inp_act[:,t,:] + state @ Wres^T)
//                  state' = (1-LEAK)*state + LEAK*new ; states_bf[:,t,:] = state'
// state is [8,1024]; WMMA uses M=16 pad (rows 8..15 zero).  The bf16 copy of
// the state is DMA'd into LDS by the Tensor Data Mover (TDM) when available.
// Launch <<<2, 1024>>>: 2 blocks x 32 waves, each wave one 16x16 N-tile.
// ---------------------------------------------------------------------------
__global__ void __launch_bounds__(1024) k_reservoir_step(
    const float* __restrict__ state_in, const bf16* __restrict__ state_in_bf,
    float* __restrict__ state_out, bf16* __restrict__ state_out_bf,
    const bf16* __restrict__ Wres, const float* __restrict__ inp_act,
    bf16* __restrict__ states_bf, int t)
{
  __shared__ bf16 sst[B_ * R_];   // 16 KB

#if __has_builtin(__builtin_amdgcn_tensor_load_to_lds) && \
    __has_builtin(__builtin_amdgcn_s_wait_tensorcnt)
  if (threadIdx.x < 32) {   // one wave issues the TDM descriptor (EXEC ignored)
    const unsigned long long ga = (unsigned long long)(uintptr_t)state_in_bf;
    const unsigned lds_off = (unsigned)(uintptr_t)(&sst[0]);
    // D# group 0: count=1, lds_addr, global_addr[56:0], type=2 ("image")
    v4u g0;
    g0[0] = 1u;                                  // count=1, user mode
    g0[1] = lds_off;                             // LDS byte address
    g0[2] = (unsigned)(ga & 0xFFFFFFFFull);      // global_addr[31:0]
    g0[3] = (unsigned)((ga >> 32) & 0x1FFFFFFull) | (2u << 30);  // [56:32] | type=2
    // D# group 1: data_size=2B, tensor 1024x8, tile 1024x8, stride0=1024
    v8i g1;
    g1[0] = (1 << 16);                           // data_size=1 -> 2 bytes
    g1[1] = (int)(1024u << 16);                  // tensor_dim0[15:0] @ bits[31:16]
    g1[2] = (int)(8u << 16);                     // dim0 hi=0 | tensor_dim1[15:0]
    g1[3] = (int)(1024u << 16);                  // dim1 hi=0 | tile_dim0=1024
    g1[4] = 8;                                   // tile_dim1=8, tile_dim2 unused
    g1[5] = 1024;                                // tensor_dim0_stride[31:0]
    g1[6] = 0;                                   // stride0 hi | stride1 lo
    g1[7] = 0;
    v4i gz4 = {};                                // groups 2/3 unused (2D tensor)
    v8i gz8 = {};                                // extra group (clang-23 6-arg form)
    __builtin_amdgcn_tensor_load_to_lds(g0, g1, gz4, gz4, gz8, 0);
    __builtin_amdgcn_s_wait_tensorcnt(0);
  }
  __syncthreads();
#else
  for (int i = threadIdx.x; i < B_ * R_; i += blockDim.x)
    sst[i] = state_in_bf[i];
  __syncthreads();
#endif

  const int lane = threadIdx.x & 31;
  const int wave = threadIdx.x >> 5;
  const int half = lane >> 4;
  const int lr   = lane & 15;
  const int n0 = (blockIdx.x * 32 + wave) * 16;
  const bf16* brow = Wres + (long long)(n0 + lr) * R_;

  v8f c = {};
  for (int k = 0; k < R_; k += 32) {
    v16bf av;
    if (lr < B_) {
      v8bf alo = *(const v8bf*)(sst + lr * R_ + k + half * 16);
      v8bf ahi = *(const v8bf*)(sst + lr * R_ + k + half * 16 + 8);
#pragma unroll
      for (int i = 0; i < 8; i++) { av[i] = alo[i]; av[i + 8] = ahi[i]; }
    } else {
#pragma unroll
      for (int i = 0; i < 16; i++) av[i] = (bf16)0.0f;
    }
    v16bf bv = *(const v16bf*)(brow + k + half * 16);
    c = wmma_bf16x32(av, bv, c);
  }

#pragma unroll
  for (int r = 0; r < 8; r++) {
    const int b = half * 8 + r;                    // batch row (pad rows skipped)
    if (b < B_) {
      const int n = n0 + lr;
      const float u  = inp_act[((long long)b * T_ + t) * R_ + n] + c[r];
      const float nw = tanhf(u);
      const float st = (1.0f - LEAK_) * state_in[b * R_ + n] + LEAK_ * nw;
      state_out[b * R_ + n] = st;
      const bf16 sb = (bf16)st;
      state_out_bf[b * R_ + n] = sb;
      states_bf[((long long)b * T_ + t) * R_ + n] = sb;
    }
  }
}

// ---------------------------------------------------------------------------
// LayerNorm over E=512: one block (256 threads) per row, writes bf16
// ---------------------------------------------------------------------------
__global__ void __launch_bounds__(256) k_layernorm(
    const float* __restrict__ x, const float* __restrict__ g,
    const float* __restrict__ b, bf16* __restrict__ y)
{
  __shared__ float red[256];
  const int row = blockIdx.x;
  const int tid = threadIdx.x;
  const float* xr = x + (long long)row * E_;
  float x0 = xr[tid], x1 = xr[tid + 256];
  red[tid] = x0 + x1; __syncthreads();
  for (int s = 128; s > 0; s >>= 1) { if (tid < s) red[tid] += red[tid + s]; __syncthreads(); }
  const float mean = red[0] * (1.0f / E_);
  __syncthreads();
  const float d0 = x0 - mean, d1 = x1 - mean;
  red[tid] = d0 * d0 + d1 * d1; __syncthreads();
  for (int s = 128; s > 0; s >>= 1) { if (tid < s) red[tid] += red[tid + s]; __syncthreads(); }
  const float rstd = rsqrtf(red[0] * (1.0f / E_) + 1e-5f);
  bf16* yr = y + (long long)row * E_;
  yr[tid]       = (bf16)(d0 * rstd * g[tid] + b[tid]);
  yr[tid + 256] = (bf16)(d1 * rstd * g[tid + 256] + b[tid + 256]);
}

// ---------------------------------------------------------------------------
// QKV reshape:  qkv f32 [B*T, 3E]  ->  q,k bf16 [B,H,T,DH], vT bf16 [B,H,DH,T]
// ---------------------------------------------------------------------------
__global__ void k_qkv_reshape(const float* __restrict__ qkv,
                              bf16* __restrict__ q, bf16* __restrict__ kk,
                              bf16* __restrict__ vt)
{
  const int g = blockIdx.x * blockDim.x + threadIdx.x;   // ((b*T+t)*H+h)*DH+d
  const int d = g & (DH_ - 1);
  const int h = (g >> 6) & (H_ - 1);
  const int t = (g >> 9) & (T_ - 1);
  const int b = g >> 19;
  const long long row = (long long)b * T_ + t;
  const float* src = qkv + row * (3 * E_);
  const int col = h * DH_ + d;
  const long long qi = (((long long)b * H_ + h) * T_ + t) * DH_ + d;
  q[qi]  = (bf16)src[col];
  kk[qi] = (bf16)src[E_ + col];
  vt[(((long long)b * H_ + h) * DH_ + d) * T_ + t] = (bf16)src[2 * E_ + col];
}

// ---------------------------------------------------------------------------
// Softmax with phase-distance bias.  scores f32 [H,T,T] for one batch b.
//   attn[t,s] = softmax_s( scores*scale + exp(-|phase[t]-phase[s]|/temp) )
// grid = (T, H), block = 256 (4 columns each)
// ---------------------------------------------------------------------------
__global__ void __launch_bounds__(256) k_softmax_bias(
    const float* __restrict__ scores, bf16* __restrict__ attn,
    const float* __restrict__ phase, const float* __restrict__ temps,
    int l, int b)
{
  __shared__ float red[256];
  const int t = blockIdx.x;
  const int h = blockIdx.y;
  const int tid = threadIdx.x;
  const float temp  = temps[l];
  const float scale = 0.125f;          // 1/sqrt(DH=64)
  const float* ph = phase + ((long long)b * T_) * P_;
  const float p0 = ph[t * P_ + 0], p1 = ph[t * P_ + 1], p2 = ph[t * P_ + 2];
  const float* srow = scores + ((long long)h * T_ + t) * T_;

  float vals[4];
  float lm = -3.4e38f;
#pragma unroll
  for (int j = 0; j < 4; j++) {
    const int s = tid + j * 256;
    const float q0 = ph[s * P_ + 0] - p0;
    const float q1 = ph[s * P_ + 1] - p1;
    const float q2 = ph[s * P_ + 2] - p2;
    const float dist = sqrtf(fmaxf(q0 * q0 + q1 * q1 + q2 * q2, 0.0f));
    const float v = srow[s] * scale + expf(-dist / temp);
    vals[j] = v;
    lm = fmaxf(lm, v);
  }
  red[tid] = lm; __syncthreads();
  for (int s = 128; s > 0; s >>= 1) { if (tid < s) red[tid] = fmaxf(red[tid], red[tid + s]); __syncthreads(); }
  const float m = red[0]; __syncthreads();
  float ls = 0.0f;
#pragma unroll
  for (int j = 0; j < 4; j++) { vals[j] = expf(vals[j] - m); ls += vals[j]; }
  red[tid] = ls; __syncthreads();
  for (int s = 128; s > 0; s >>= 1) { if (tid < s) red[tid] += red[tid + s]; __syncthreads(); }
  const float inv = 1.0f / red[0];
  bf16* arow = attn + ((long long)h * T_ + t) * T_;
#pragma unroll
  for (int j = 0; j < 4; j++) arow[tid + j * 256] = (bf16)(vals[j] * inv);
}

// o_bh f32 [B,H,T,DH] -> o_bt bf16 [B*T, E]
__global__ void k_o_reshape(const float* __restrict__ obh, bf16* __restrict__ obt)
{
  const int g = blockIdx.x * blockDim.x + threadIdx.x;
  const int d = g & 63;
  const int h = (g >> 6) & 7;
  const int t = (g >> 9) & 1023;
  const int b = g >> 19;
  obt[((long long)b * T_ + t) * E_ + h * DH_ + d] =
      (bf16)obh[(((long long)b * H_ + h) * T_ + t) * DH_ + d];
}

// ---------------------------------------------------------------------------
// Heads (tiny): hh = relu(last @ W1^T + b1); p = hh @ W2^T + b2 (+sigmoid for 2,3)
// ---------------------------------------------------------------------------
__global__ void k_heads1(const float* __restrict__ hidden, const float* __restrict__ W1,
                         const float* __restrict__ b1, float* __restrict__ hh)
{
  const int g = blockIdx.x * blockDim.x + threadIdx.x;  // (i*8+b)*64+j
  if (g >= 4 * B_ * 64) return;
  const int j = g & 63;
  const int b = (g >> 6) & 7;
  const int i = g >> 9;
  const float* xr = hidden + ((long long)b * T_ + (T_ - 1)) * E_;
  const float* wr = W1 + ((long long)i * 64 + j) * E_;
  float s = b1[i * 64 + j];
  for (int k = 0; k < E_; k++) s += xr[k] * wr[k];
  hh[g] = fmaxf(s, 0.0f);
}

__global__ void k_heads2(const float* __restrict__ hh, const float* __restrict__ W2,
                         const float* __restrict__ b2, float* __restrict__ out)
{
  const int tid = threadIdx.x;
  if (tid >= 32) return;
  const int i = tid >> 3, b = tid & 7;
  const float* hr = hh + (i * 8 + b) * 64;
  const float* wr = W2 + i * 64;
  float s = b2[i];
  for (int j = 0; j < 64; j++) s += hr[j] * wr[j];
  if (i >= 2) s = 1.0f / (1.0f + expf(-s));
  out[i * 8 + b] = s;
}

// ---------------------------------------------------------------------------
// Host side
// ---------------------------------------------------------------------------
static inline void launch_gemm(const bf16* A, const bf16* Bt, const float* bias, void* C,
                               int M, int N, int K,
                               long long sA, long long sB, long long sC, int batch,
                               int act, int outbf, hipStream_t st)
{
  dim3 grid((N + 127) / 128, (M + 127) / 128, batch);
  k_gemm<<<grid, 256, 0, st>>>(A, Bt, bias, C, M, N, K, sA, sB, sC, act, outbf);
}

static inline void launch_conv(const float* x, bf16* y, long long n, hipStream_t st) {
  k_f32_to_bf16<<<(unsigned)((n + 255) / 256), 256, 0, st>>>(x, y, (int)n);
}

extern "C" void kernel_launch(void* const* d_in, const int* in_sizes, int n_in,
                              void* d_out, int out_size, void* d_ws, size_t ws_size,
                              hipStream_t stream)
{
  (void)in_sizes; (void)n_in; (void)out_size; (void)ws_size;
  // ---- inputs (setup_inputs order) ----
  const float* x        = (const float*)d_in[0];   // [8,1024,64]
  const float* W_in     = (const float*)d_in[1];   // [1024,64]
  const float* W_res    = (const float*)d_in[2];   // [1024,1024]
  const float* emb_W    = (const float*)d_in[3];   // [512,1024]
  const float* emb_b    = (const float*)d_in[4];   // [512]
  const float* ph_W1    = (const float*)d_in[5];   // [64,512]
  const float* ph_b1    = (const float*)d_in[6];   // [64]
  const float* ph_W2    = (const float*)d_in[7];   // [3,64]
  const float* ph_b2    = (const float*)d_in[8];   // [3]
  const float* qkv_W    = (const float*)d_in[9];   // [4,1536,512]
  const float* qkv_b    = (const float*)d_in[10];  // [4,1536]
  const float* out_W    = (const float*)d_in[11];  // [4,512,512]
  const float* out_b    = (const float*)d_in[12];  // [4,512]
  const float* ph_temp  = (const float*)d_in[13];  // [4]
  const float* ln1_g    = (const float*)d_in[14];
  const float* ln1_b    = (const float*)d_in[15];
  const float* ffn_W1   = (const float*)d_in[16];  // [4,2048,512]
  const float* ffn_b1   = (const float*)d_in[17];  // [4,2048]
  const float* ffn_W2   = (const float*)d_in[18];  // [4,512,2048]
  const float* ffn_b2   = (const float*)d_in[19];  // [4,512]
  const float* ln2_g    = (const float*)d_in[20];
  const float* ln2_b    = (const float*)d_in[21];
  const float* heads_W1 = (const float*)d_in[22];  // [4,64,512]
  const float* heads_b1 = (const float*)d_in[23];  // [4,64]
  const float* heads_W2 = (const float*)d_in[24];  // [4,1,64]
  const float* heads_b2 = (const float*)d_in[25];  // [4,1]
  float* out = (float*)d_out;

  // ---- workspace carve (256B aligned chunks) ----
  char* p = (char*)d_ws;
  auto take = [&](size_t bytes) -> void* {
    void* r = (void*)p;
    p += (bytes + 255) & ~(size_t)255;
    return r;
  };
  const long long MT = (long long)B_ * T_;  // 8192 rows
  bf16* Win_b  = (bf16*)take((size_t)R_ * DIN_ * 2);
  bf16* Wres_b = (bf16*)take((size_t)R_ * R_ * 2);
  bf16* embW_b = (bf16*)take((size_t)E_ * R_ * 2);
  bf16* phW1_b = (bf16*)take((size_t)64 * E_ * 2);
  bf16* phW2_b = (bf16*)take((size_t)P_ * 64 * 2);
  bf16* qkvW_b = (bf16*)take((size_t)L_ * 3 * E_ * E_ * 2);
  bf16* outW_b = (bf16*)take((size_t)L_ * E_ * E_ * 2);
  bf16* fW1_b  = (bf16*)take((size_t)L_ * 4 * E_ * E_ * 2);
  bf16* fW2_b  = (bf16*)take((size_t)L_ * E_ * 4 * E_ * 2);
  bf16* x_b    = (bf16*)take((size_t)MT * DIN_ * 2);
  float* inp_act = (float*)take((size_t)MT * R_ * 4);
  float* stateA  = (float*)take((size_t)B_ * R_ * 4);
  float* stateB  = (float*)take((size_t)B_ * R_ * 4);
  bf16*  stateBfA= (bf16*)take((size_t)B_ * R_ * 2);
  bf16*  stateBfB= (bf16*)take((size_t)B_ * R_ * 2);
  bf16*  states_b= (bf16*)take((size_t)MT * R_ * 2);
  float* hidden  = (float*)take((size_t)MT * E_ * 4);
  bf16*  emb_b16 = (bf16*)take((size_t)MT * E_ * 2);
  bf16*  ph_h    = (bf16*)take((size_t)MT * 64 * 2);
  float* phase   = (float*)take((size_t)MT * P_ * 4);
  bf16*  xn      = (bf16*)take((size_t)MT * E_ * 2);
  float* qkvf    = (float*)take((size_t)MT * 3 * E_ * 4);
  bf16*  q_b     = (bf16*)take((size_t)B_ * H_ * T_ * DH_ * 2);
  bf16*  k_bm    = (bf16*)take((size_t)B_ * H_ * T_ * DH_ * 2);
  bf16*  vt_b    = (bf16*)take((size_t)B_ * H_ * DH_ * T_ * 2);
  float* scores  = (float*)take((size_t)H_ * T_ * T_ * 4);   // per-b reuse
  bf16*  attn    = (bf16*)take((size_t)H_ * T_ * T_ * 2);    // per-b reuse
  float* o_bh    = (float*)take((size_t)B_ * H_ * T_ * DH_ * 4);
  bf16*  o_bt    = (bf16*)take((size_t)MT * E_ * 2);
  float* tmp     = (float*)take((size_t)MT * E_ * 4);
  bf16*  ffn_h   = (bf16*)take((size_t)MT * 4 * E_ * 2);
  float* hh      = (float*)take((size_t)4 * B_ * 64 * 4);

  // ---- weight / input conversion to bf16 ----
  launch_conv(x,      x_b,    MT * DIN_, stream);
  launch_conv(W_in,   Win_b,  (long long)R_ * DIN_, stream);
  launch_conv(W_res,  Wres_b, (long long)R_ * R_, stream);
  launch_conv(emb_W,  embW_b, (long long)E_ * R_, stream);
  launch_conv(ph_W1,  phW1_b, (long long)64 * E_, stream);
  launch_conv(ph_W2,  phW2_b, (long long)P_ * 64, stream);
  launch_conv(qkv_W,  qkvW_b, (long long)L_ * 3 * E_ * E_, stream);
  launch_conv(out_W,  outW_b, (long long)L_ * E_ * E_, stream);
  launch_conv(ffn_W1, fW1_b,  (long long)L_ * 4 * E_ * E_, stream);
  launch_conv(ffn_W2, fW2_b,  (long long)L_ * E_ * 4 * E_, stream);

  // ---- inp_act = x @ W_in^T : [8192,64] x [1024,64]^T ----
  launch_gemm(x_b, Win_b, nullptr, inp_act, (int)MT, R_, DIN_, 0, 0, 0, 1, 0, 0, stream);

  // ---- reservoir scan (sequential, WMMA + TDM per step, ping-pong state) ----
  k_fill<<<(B_ * R_ + 255) / 256, 256, 0, stream>>>(stateA, B_ * R_, 0.0f);
  k_fill_bf16<<<(B_ * R_ + 255) / 256, 256, 0, stream>>>(stateBfA, B_ * R_, 0.0f);
  for (int t = 0; t < T_; t++) {
    const float* sin  = (t & 1) ? stateB : stateA;
    const bf16*  sinb = (t & 1) ? stateBfB : stateBfA;
    float* sout       = (t & 1) ? stateA : stateB;
    bf16*  soutb      = (t & 1) ? stateBfA : stateBfB;
    k_reservoir_step<<<2, 1024, 0, stream>>>(sin, sinb, sout, soutb,
                                             Wres_b, inp_act, states_b, t);
  }

  // ---- emb = states @ emb_W^T + emb_b  (-> hidden f32) ----
  launch_gemm(states_b, embW_b, emb_b, hidden, (int)MT, E_, R_, 0, 0, 0, 1, 0, 0, stream);
  launch_conv(hidden, emb_b16, MT * E_, stream);

  // ---- phase head: tanh(emb @ phW1^T + b1) @ phW2^T + b2 ----
  launch_gemm(emb_b16, phW1_b, ph_b1, ph_h, (int)MT, 64, E_, 0, 0, 0, 1, /*tanh*/1, /*bf16*/1, stream);
  launch_gemm(ph_h, phW2_b, ph_b2, phase, (int)MT, P_, 64, 0, 0, 0, 1, 0, 0, stream);
  k_copy<<<(int)((MT * P_ + 255) / 256), 256, 0, stream>>>(phase, out + 32, (int)(MT * P_));

  // ---- transformer layers ----
  const long long sQK = (long long)T_ * DH_;
  const long long sS  = (long long)T_ * T_;
  for (int l = 0; l < L_; l++) {
    k_layernorm<<<(unsigned)MT, 256, 0, stream>>>(hidden, ln1_g + l * E_, ln1_b + l * E_, xn);
    launch_gemm(xn, qkvW_b + (size_t)l * 3 * E_ * E_, qkv_b + l * 3 * E_, qkvf,
                (int)MT, 3 * E_, E_, 0, 0, 0, 1, 0, 0, stream);
    k_qkv_reshape<<<(unsigned)(B_ * T_ * H_ * DH_ / 256), 256, 0, stream>>>(qkvf, q_b, k_bm, vt_b);

    for (int b = 0; b < B_; b++) {
      const long long off = (long long)b * H_ * T_ * DH_;
      // scores[h] = Q[b,h] @ K[b,h]^T   (K rows are already [s, d] == Bt layout)
      launch_gemm(q_b + off, k_bm + off, nullptr, scores,
                  T_, T_, DH_, sQK, sQK, sS, H_, 0, 0, stream);
      k_softmax_bias<<<dim3(T_, H_), 256, 0, stream>>>(scores, attn, phase, ph_temp, l, b);
      // o[h] = attn[h] @ V[b,h]   (Bt = V^T [d, s])
      launch_gemm(attn, vt_b + (long long)b * H_ * DH_ * T_, nullptr, o_bh + off,
                  T_, DH_, T_, sS, (long long)DH_ * T_, sQK, H_, 0, 0, stream);
    }
    k_o_reshape<<<(unsigned)(B_ * T_ * H_ * DH_ / 256), 256, 0, stream>>>(o_bh, o_bt);

    launch_gemm(o_bt, outW_b + (size_t)l * E_ * E_, out_b + l * E_, tmp,
                (int)MT, E_, E_, 0, 0, 0, 1, 0, 0, stream);
    k_add<<<(unsigned)((MT * E_) / 256), 256, 0, stream>>>(hidden, tmp, (int)(MT * E_));

    k_layernorm<<<(unsigned)MT, 256, 0, stream>>>(hidden, ln2_g + l * E_, ln2_b + l * E_, xn);
    launch_gemm(xn, fW1_b + (size_t)l * 4 * E_ * E_, ffn_b1 + l * 4 * E_, ffn_h,
                (int)MT, 4 * E_, E_, 0, 0, 0, 1, /*gelu*/2, /*bf16*/1, stream);
    launch_gemm(ffn_h, fW2_b + (size_t)l * E_ * 4 * E_, ffn_b2 + l * E_, tmp,
                (int)MT, E_, 4 * E_, 0, 0, 0, 1, 0, 0, stream);
    k_add<<<(unsigned)((MT * E_) / 256), 256, 0, stream>>>(hidden, tmp, (int)(MT * E_));
  }

  // ---- prediction heads ----
  k_heads1<<<(4 * B_ * 64 + 255) / 256, 256, 0, stream>>>(hidden, heads_W1, heads_b1, hh);
  k_heads2<<<1, 32, 0, stream>>>(hh, heads_W2, heads_b2, out);
}
